// SubSampler_9921374453826
// MI455X (gfx1250) — matrix operand order repaired and checked by reading, not compiled, not run on hardware
//
#include <hip/hip_runtime.h>
#include <hip/hip_bf16.h>

#define BSZ   4
#define DCH   128
#define HIMG  256
#define WIMG  256
#define BORD  16
#define SUBS  4
#define NS    56              // number of subsampled coords per axis
#define NPTS  (NS * NS)       // 3136 = 196 * 16
#define NT    (NPTS / 16)     // 196 tiles of 16
#define KCH   4               // 128 / 32 K-chunks
#define KTPW  4               // k-tiles per wave (A-fragment reuse)
#define NGRP  (NT / KTPW)     // 49 k-tile groups

typedef __bf16 bf16;
typedef __attribute__((ext_vector_type(16))) __bf16 bf16x16;
typedef __attribute__((ext_vector_type(8)))  float  v8f;

// ---------------------------------------------------------------------------
// Kernel 1: bilinear warp of feat2/conf2 at the 56x56 subsampled points.
// One block per (b, point); thread d handles channel d. Writes bf16 Kdense
// row-major [b][k][d] so the GEMM's B-fragment is a contiguous v16bf load.
// Also produces m2 (mask) and qconf outputs.
// ---------------------------------------------------------------------------
__global__ __launch_bounds__(128)
void warp_kernel(const float* __restrict__ feat2,
                 const float* __restrict__ conf1,
                 const float* __restrict__ conf2,
                 const float* __restrict__ aflow,
                 bf16* __restrict__ Kdense,
                 float* __restrict__ m2_out,
                 float* __restrict__ qconf_out) {
  const int bp = blockIdx.x;
  const int b  = bp / NPTS;
  const int k  = bp % NPTS;
  const int ky = k / NS, kx = k % NS;
  const int y  = BORD + SUBS * ky;
  const int x  = BORD + SUBS * kx;
  const long pix = (long)y * WIMG + x;

  // pixel coords from aflow (NaN -> 9e9 like the reference)
  const long abase = (long)b * 2 * HIMG * WIMG;
  float ax = aflow[abase + pix];
  float ay = aflow[abase + (long)HIMG * WIMG + pix];
  float gx = ax * (2.0f / (WIMG - 1)) - 1.0f;
  float gy = ay * (2.0f / (HIMG - 1)) - 1.0f;
  gx = (gx != gx) ? 9.0e9f : gx;
  gy = (gy != gy) ? 9.0e9f : gy;
  const float px = ((gx + 1.0f) * WIMG - 1.0f) * 0.5f;
  const float py = ((gy + 1.0f) * HIMG - 1.0f) * 0.5f;

  const float x0f = floorf(px), y0f = floorf(py);
  const float wx = px - x0f, wy = py - y0f;
  // validity computed in float domain (avoids int overflow on 9e9)
  const float x1f = x0f + 1.0f, y1f = y0f + 1.0f;
  const float vx0 = (x0f >= 0.0f && x0f < (float)WIMG) ? 1.0f : 0.0f;
  const float vx1 = (x1f >= 0.0f && x1f < (float)WIMG) ? 1.0f : 0.0f;
  const float vy0 = (y0f >= 0.0f && y0f < (float)HIMG) ? 1.0f : 0.0f;
  const float vy1 = (y1f >= 0.0f && y1f < (float)HIMG) ? 1.0f : 0.0f;
  const int x0c = (int)fminf(fmaxf(x0f, 0.0f), (float)(WIMG - 1));
  const int x1c = (int)fminf(fmaxf(x1f, 0.0f), (float)(WIMG - 1));
  const int y0c = (int)fminf(fmaxf(y0f, 0.0f), (float)(HIMG - 1));
  const int y1c = (int)fminf(fmaxf(y1f, 0.0f), (float)(HIMG - 1));
  const long o00 = (long)y0c * WIMG + x0c;
  const long o01 = (long)y0c * WIMG + x1c;
  const long o10 = (long)y1c * WIMG + x0c;
  const long o11 = (long)y1c * WIMG + x1c;
  const float w00 = (1.0f - wy) * (1.0f - wx) * vy0 * vx0;
  const float w01 = (1.0f - wy) * wx          * vy0 * vx1;
  const float w10 = wy          * (1.0f - wx) * vy1 * vx0;
  const float w11 = wy          * wx          * vy1 * vx1;

  const int d = threadIdx.x;
  const float* f2 = feat2 + ((long)b * DCH + d) * (long)HIMG * WIMG;
  const float val = w00 * f2[o00] + w01 * f2[o01] + w10 * f2[o10] + w11 * f2[o11];
  Kdense[((long)b * NPTS + k) * DCH + d] = (bf16)val;

  if (d == 0) {
    const float* c2 = conf2 + (long)b * HIMG * WIMG;
    const float c2w = w00 * c2[o00] + w01 * c2[o01] + w10 * c2[o10] + w11 * c2[o11];
    const float c1v = conf1[(long)b * HIMG * WIMG + pix];
    // mask: rounded coords in-bounds (rintf == round-half-even like jnp.round)
    const float rx = rintf(px), ry = rintf(py);
    const bool inb = (ry >= 0.0f) && (ry < (float)HIMG) && (rx >= 0.0f) && (rx < (float)WIMG);
    m2_out[(long)b * NPTS + k]    = inb ? 1.0f : 0.0f;
    qconf_out[(long)b * NPTS + k] = 0.5f * (c1v + c2w);
  }
}

// ---------------------------------------------------------------------------
// Kernel 2: gather feat1 at query points directly into the WMMA 16-bit
// A-fragment layout (16x32, M x K), in this file's contiguous k order (same as B):
//   lane L: m = L%16, h = L/16
//   half j: K = h*16 + j
// One thread per (b, qtile, kchunk, lane); writes 16 contiguous bf16.
// ---------------------------------------------------------------------------
__global__ __launch_bounds__(256)
void apack_kernel(const float* __restrict__ feat1, bf16* __restrict__ Afrag) {
  const int t = blockIdx.x * 256 + threadIdx.x;
  if (t >= BSZ * NT * KCH * 32) return;
  const int lane = t & 31;
  const int c    = (t >> 5) & 3;
  const int qt   = (t >> 7) % NT;
  const int b    = (t >> 7) / NT;

  const int m = lane & 15, h = lane >> 4;
  const int q  = qt * 16 + m;
  const int qy = q / NS, qx = q % NS;
  const int y  = BORD + SUBS * qy;
  const int x  = BORD + SUBS * qx;
  const float* base = feat1 + (long)b * DCH * HIMG * WIMG + (long)y * WIMG + x;

  bf16* out = Afrag + (size_t)t * 16;
#pragma unroll
  for (int j = 0; j < 16; ++j) {
    const int v = j >> 1, p = j & 1;
    const int kloc = h * 16 + v * 2 + p;
    const int kd   = c * 32 + kloc;
    out[j] = (bf16)base[(long)kd * HIMG * WIMG];
  }
}

// ---------------------------------------------------------------------------
// Kernel 3: scores[b,q,k] = sum_d Q[d,q] * K[d,k] via v_wmma_f32_16x16x32_bf16.
// 8 waves / block; each wave owns one qtile x 4 consecutive ktiles, so the
// A-fragment registers are reused 4x per load (16 WMMAs per wave).
// A: pre-swizzled fragment (contiguous 32 B / lane).
// B: straight from Kdense — lane n=L%16 reads K-range h*16.. (contiguous 32 B).
// C store per ISA layout: VGPR r -> M = r + 8*h, N = lane%16.
// scores/gt are write-once streaming outputs (~315 MB) -> non-temporal stores
// keep L2 free for the 6.4 MB of packed operands every tile re-reads.
// ---------------------------------------------------------------------------
__global__ __launch_bounds__(256)
void gemm_kernel(const bf16* __restrict__ Afrag,
                 const bf16* __restrict__ Kdense,
                 float* __restrict__ scores,
                 float* __restrict__ gt) {
  const int lane = threadIdx.x & 31;
  const int wv   = threadIdx.x >> 5;
  const int g    = blockIdx.x * 8 + wv;   // uniform per wave -> EXEC stays all-1s
  const int qt   = blockIdx.y;
  const int b    = blockIdx.z;
  if (g >= NGRP) return;                  // whole-wave exit only
  const int kt0 = g * KTPW;

  const int n = lane & 15, h = lane >> 4;
  const bf16* aptr  = Afrag  + (((size_t)b * NT + qt) * KCH * 32 + lane) * 16;
  const bf16* bbase = Kdense + ((size_t)b * NPTS + (size_t)kt0 * 16 + n) * DCH + h * 16;

  v8f acc[KTPW];
#pragma unroll
  for (int t = 0; t < KTPW; ++t) acc[t] = (v8f){};

#pragma unroll
  for (int c = 0; c < KCH; ++c) {
    const bf16x16 a = *(const bf16x16*)(aptr + (size_t)c * 32 * 16);
#pragma unroll
    for (int t = 0; t < KTPW; ++t) {
      const bf16x16 bf = *(const bf16x16*)(bbase + (size_t)t * 16 * DCH + c * 32);
      acc[t] = __builtin_amdgcn_wmma_f32_16x16x32_bf16(
          /*neg_a=*/false, a, /*neg_b=*/false, bf,
          /*c_mod=*/(short)0, acc[t], /*reuse_a=*/false, /*reuse_b=*/false);
    }
  }

  const int qg = qt * 16;
  const size_t orow = ((size_t)b * NPTS + qg) * NPTS + (size_t)kt0 * 16 + n;
#pragma unroll
  for (int t = 0; t < KTPW; ++t) {
    const int kg = (kt0 + t) * 16 + n;
#pragma unroll
    for (int r = 0; r < 8; ++r) {
      const int m = r + 8 * h;
      const size_t off = orow + (size_t)m * NPTS + (size_t)t * 16;
      __builtin_nontemporal_store(acc[t][r], scores + off);
      __builtin_nontemporal_store((qg + m == kg) ? 1.0f : 0.0f, gt + off);
    }
  }
}

// ---------------------------------------------------------------------------
extern "C" void kernel_launch(void* const* d_in, const int* in_sizes, int n_in,
                              void* d_out, int out_size, void* d_ws, size_t ws_size,
                              hipStream_t stream) {
  const float* feat1 = (const float*)d_in[0];
  const float* feat2 = (const float*)d_in[1];
  const float* conf1 = (const float*)d_in[2];
  const float* conf2 = (const float*)d_in[3];
  const float* aflow = (const float*)d_in[4];

  float* out = (float*)d_out;
  const size_t SC = (size_t)BSZ * NPTS * NPTS;   // scores elements
  float* scores = out;
  float* gtout  = out + SC;
  float* m2out  = out + 2 * SC;
  float* qconf  = out + 2 * SC + (size_t)BSZ * NPTS;

  // workspace: Kdense (bf16 [B][NPTS][D]) then Afrag (bf16 [B][NT][KCH][32][16])
  bf16* Kdense = (bf16*)d_ws;
  const size_t kdense_bytes = (size_t)BSZ * NPTS * DCH * sizeof(bf16);  // 3.2 MB
  bf16* Afrag = (bf16*)((char*)d_ws + ((kdense_bytes + 255) & ~(size_t)255));

  // 1) warp feat2/conf2 -> Kdense (bf16), m2, qconf
  warp_kernel<<<BSZ * NPTS, 128, 0, stream>>>(feat2, conf1, conf2, aflow,
                                              Kdense, m2out, qconf);

  // 2) pack feat1 queries into WMMA A-fragment layout
  const int apack_threads = BSZ * NT * KCH * 32;   // 100352
  apack_kernel<<<(apack_threads + 255) / 256, 256, 0, stream>>>(feat1, Afrag);

  // 3) tiled bf16 WMMA GEMM -> scores (+ identity gt), NT streaming stores
  dim3 grid((NGRP + 7) / 8, NT, BSZ);
  gemm_kernel<<<grid, 256, 0, stream>>>(Afrag, Kdense, scores, gtout);
}
